// GCN_45268955300496
// MI455X (gfx1250) — compile-verified, matrix operand order checked by reading, not run on hardware
//
#include <hip/hip_runtime.h>
#include <hip/hip_fp16.h>
#include <math.h>

typedef __attribute__((ext_vector_type(16))) _Float16 v16h;
typedef __attribute__((ext_vector_type(8)))  _Float16 v8h;
typedef __attribute__((ext_vector_type(8)))  float    v8f;

#define NFEAT 128
#define LDSW  136   // padded W row stride in halves (+16B to spread LDS banks)

// ---------------- degree / normalization ----------------

__global__ void gcn_deg_init(float* deg, int n) {
  int i = blockIdx.x * blockDim.x + threadIdx.x;
  if (i < n) deg[i] = 1.0f;                       // self-loop
}

__global__ void gcn_deg_count(const int* __restrict__ col, float* deg, int E) {
  int e = blockIdx.x * blockDim.x + threadIdx.x;
  if (e < E)
    __hip_atomic_fetch_add(&deg[col[e]], 1.0f, __ATOMIC_RELAXED, __HIP_MEMORY_SCOPE_AGENT);
}

__global__ void gcn_dinv(float* deg, int n) {     // in-place: deg -> rsqrt(deg)
  int i = blockIdx.x * blockDim.x + threadIdx.x;
  if (i < n) deg[i] = rsqrtf(deg[i]);
}

// ---------------- fused GEMM + pre/post normalization ----------------
// Computes: hs[m,:] = dinv[m] * ( f(A[m,:]) @ W ),  agg[m,:] = hs[m,:]  (self-loop init)
// APPLY_IN: f(a)[k] = relu(dinv[m]*a[k] + bias_in[k])   (layer-2 input transform)
// else     f(a) = a                                      (layer-1: raw X)
template<bool APPLY_IN>
__global__ __launch_bounds__(256)
void gcn_gemm(const float* A,                      // [n,128] (may alias agg for layer 2)
              const float* __restrict__ W,         // [128,128] row-major K x N
              const float* __restrict__ dinv,      // [n]
              const float* __restrict__ bias_in,   // [128] or nullptr
              float* hs, float* agg, int n)
{
  __shared__ _Float16 Wt[NFEAT][LDSW];            // W transposed: Wt[n][k], f16
  __shared__ float    bsh[NFEAT];

  const int tid = threadIdx.x;
  for (int idx = tid; idx < NFEAT * NFEAT; idx += 256) {
    int k = idx >> 7, nn = idx & 127;
    Wt[nn][k] = (_Float16)W[idx];
  }
  if (APPLY_IN) {
    for (int idx = tid; idx < NFEAT; idx += 256) bsh[idx] = bias_in[idx];
  }
  __syncthreads();

  const int wave = tid >> 5;
  const int lane = tid & 31;
  const int l16  = lane & 15;
  const int kg   = lane >> 4;                     // K-group per ISA 16-bit A/B layout
  const int rbase = blockIdx.x * 128 + wave * 16; // 16-row tile per wave
  const int row   = rbase + l16;                  // A row this lane loads
  const bool rv   = row < n;

  float dsc = 0.0f;
  if (APPLY_IN && rv) dsc = dinv[row];

  v8f acc[8] = {};                                // 16x16 tiles covering all 128 cols
  const float* arow = A + (long)row * NFEAT;

  for (int kc = 0; kc < 4; ++kc) {                // K = 128 in 4 steps of 32
    const int kb = kc * 32;
    const int k0 = kb + kg * 8;                   // elements 0..7  -> K = k0..k0+7
    const int k1 = kb + 16 + kg * 8;              // elements 8..15 -> K = k1..k1+7
    v16h a = {};
    if (rv) {
      #pragma unroll
      for (int e = 0; e < 8; ++e) {
        float x0 = arow[k0 + e];
        float x1 = arow[k1 + e];
        if (APPLY_IN) {
          x0 = fmaxf(dsc * x0 + bsh[k0 + e], 0.0f);
          x1 = fmaxf(dsc * x1 + bsh[k1 + e], 0.0f);
        }
        a[e]     = (_Float16)x0;
        a[8 + e] = (_Float16)x1;
      }
    }
    #pragma unroll
    for (int nt = 0; nt < 8; ++nt) {
      const _Float16* wr = &Wt[nt * 16 + l16][0];
      v8h blo = *(const v8h*)(wr + k0);           // 16B-aligned ds_load_b128
      v8h bhi = *(const v8h*)(wr + k1);
      v16h b;
      #pragma unroll
      for (int e = 0; e < 8; ++e) { b[e] = blo[e]; b[8 + e] = bhi[e]; }
      acc[nt] = __builtin_amdgcn_wmma_f32_16x16x32_f16(
          false, a, false, b, (short)0, acc[nt], false, false);
    }
  }

  // D layout: lane writes rows rbase + 8*kg + r (r=0..7), col = nt*16 + l16
  #pragma unroll
  for (int r = 0; r < 8; ++r) {
    const int m = rbase + 8 * kg + r;
    if (m < n) {
      const float ds = dinv[m];
      #pragma unroll
      for (int nt = 0; nt < 8; ++nt) {
        const float v = ds * acc[nt][r];
        const long off = (long)m * NFEAT + nt * 16 + l16;
        hs[off]  = v;
        agg[off] = v;                             // self-loop contribution
      }
    }
  }
}

// ---------------- edge scatter: agg[col] += hs[row] ----------------
// one wave per edge; lane handles 4 consecutive floats (coalesced b128 read)
__global__ __launch_bounds__(256)
void gcn_scatter(const int* __restrict__ row, const int* __restrict__ col,
                 const float* __restrict__ hs, float* __restrict__ agg, int E)
{
  long t = (long)blockIdx.x * blockDim.x + threadIdx.x;
  int e = (int)(t >> 5);
  int c = (int)(t & 31);
  if (e < E) {
    const int r = row[e], d = col[e];
    const float4 v = *(const float4*)(hs + (long)r * NFEAT + c * 4);
    float* dst = agg + (long)d * NFEAT + c * 4;
    __hip_atomic_fetch_add(dst + 0, v.x, __ATOMIC_RELAXED, __HIP_MEMORY_SCOPE_AGENT);
    __hip_atomic_fetch_add(dst + 1, v.y, __ATOMIC_RELAXED, __HIP_MEMORY_SCOPE_AGENT);
    __hip_atomic_fetch_add(dst + 2, v.z, __ATOMIC_RELAXED, __HIP_MEMORY_SCOPE_AGENT);
    __hip_atomic_fetch_add(dst + 3, v.w, __ATOMIC_RELAXED, __HIP_MEMORY_SCOPE_AGENT);
  }
}

// ---------------- final: h2 = dinv*agg2 + b2; out = sigmoid(h2 . Wc + bc) ----------------
__global__ __launch_bounds__(256)
void gcn_final(const float* __restrict__ agg, const float* __restrict__ dinv,
               const float* __restrict__ b2, const float* __restrict__ Wc,
               const float* __restrict__ bc, float* __restrict__ out, int n)
{
  const int wave = threadIdx.x >> 5;
  const int lane = threadIdx.x & 31;
  const int i = blockIdx.x * 8 + wave;            // one wave per node
  if (i >= n) return;
  const float ds = dinv[i];
  const float4 av = *(const float4*)(agg + (long)i * NFEAT + lane * 4);
  const float4 bv = *(const float4*)(b2 + lane * 4);
  const float4 wv = *(const float4*)(Wc + lane * 4);
  float p = (ds * av.x + bv.x) * wv.x + (ds * av.y + bv.y) * wv.y +
            (ds * av.z + bv.z) * wv.z + (ds * av.w + bv.w) * wv.w;
  #pragma unroll
  for (int off = 16; off > 0; off >>= 1) p += __shfl_xor(p, off, 32);
  if (lane == 0) out[i] = 1.0f / (1.0f + __expf(-(p + bc[0])));
}

// ---------------- launcher ----------------

extern "C" void kernel_launch(void* const* d_in, const int* in_sizes, int n_in,
                              void* d_out, int out_size, void* d_ws, size_t ws_size,
                              hipStream_t stream)
{
  const float* x  = (const float*)d_in[0];
  const int*   ei = (const int*)  d_in[1];
  const float* W1 = (const float*)d_in[2];
  const float* b1 = (const float*)d_in[3];
  const float* W2 = (const float*)d_in[4];
  const float* b2 = (const float*)d_in[5];
  const float* Wc = (const float*)d_in[6];
  const float* bc = (const float*)d_in[7];
  float* out = (float*)d_out;

  const int n = in_sizes[0] / NFEAT;              // 100000
  const int E = in_sizes[1] / 2;                  // 1600000
  const int* row = ei;                            // edge_index[0]
  const int* col = ei + E;                        // edge_index[1]

  // workspace: dinv [n], hs [n,128], agg [n,128]  (~103 MB)
  float* dinv = (float*)d_ws;
  size_t npad = ((size_t)n + 63) & ~(size_t)63;   // keep hs 256B-aligned
  float* hs   = dinv + npad;
  float* agg  = hs + (size_t)n * NFEAT;

  const int tb = 256;
  gcn_deg_init <<<(n + tb - 1) / tb, tb, 0, stream>>>(dinv, n);
  gcn_deg_count<<<(E + tb - 1) / tb, tb, 0, stream>>>(col, dinv, E);
  gcn_dinv     <<<(n + tb - 1) / tb, tb, 0, stream>>>(dinv, n);

  const int gblocks = (n + 127) / 128;            // 128 rows per block (8 waves x 16)
  const long swork  = (long)E * 32;
  const int sblocks = (int)((swork + tb - 1) / tb);

  // layer 1: hs = dinv * (X @ W1); agg = hs (self-loops); then edge scatter
  gcn_gemm<false><<<gblocks, 256, 0, stream>>>(x, W1, dinv, nullptr, hs, agg, n);
  gcn_scatter    <<<sblocks, tb, 0, stream>>>(row, col, hs, agg, E);

  // layer 2 (in-place on agg): A = relu(dinv*agg + b1); hs = dinv*(A @ W2); agg = hs
  gcn_gemm<true> <<<gblocks, 256, 0, stream>>>(agg, W2, dinv, b1, hs, agg, n);
  gcn_scatter    <<<sblocks, tb, 0, stream>>>(row, col, hs, agg, E);

  // classifier head fused with layer-2 post-norm + bias
  gcn_final      <<<(n + 7) / 8, 256, 0, stream>>>(agg, dinv, b2, Wc, bc, out, n);
}
